// SoftSPINN_45629732553037
// MI455X (gfx1250) — matrix operand order repaired and checked by reading, not compiled, not run on hardware
//
#include <hip/hip_runtime.h>
#include <hip/hip_bf16.h>
#include <hip/hip_fp16.h>

typedef _Float16 h16;
typedef __attribute__((ext_vector_type(16))) _Float16 v16h;
typedef __attribute__((ext_vector_type(8)))  float    v8f;
typedef __attribute__((ext_vector_type(4)))  unsigned int v4u;
typedef __attribute__((ext_vector_type(8)))  int      v8i;
typedef __attribute__((ext_vector_type(4)))  int      v4i;

typedef __attribute__((address_space(3))) h16 ls_h16;   // LDS
typedef __attribute__((address_space(1))) h16 gl_h16;   // global

#define BETA_  10.0f
#define Bsz    64
#define Ssz    48
#define Hsz    256
#define TIME   96
#define Msz    193
#define XT     (3*Hsz)   // 768
#define G4     (4*Hsz)   // 1024
#define G5     (5*Hsz)   // 1280
#define MLPD   1024
#define KCH    64        // K-chunk staged per TDM transfer (rows of Wt)
#define PANEL_H (KCH*16) // halves per staged panel (2 KB)
#define NWAVES 16

#if __has_builtin(__builtin_amdgcn_tensor_load_to_lds)
#define TDM_OK 1
#else
#define TDM_OK 0
#endif

__device__ __forceinline__ float sigm(float x){ return 1.0f/(1.0f + __expf(-x)); }

__device__ __forceinline__ unsigned rl(unsigned x){
  return (unsigned)__builtin_amdgcn_readfirstlane((int)x);
}

// ---- WMMA fragment loads per CDNA5 16x16x32 f16 layouts ----
// A (16x32, MxK): lanes 0-15 -> M=lane, halves i: K = (i<8? i : i+8);
//                 lanes 16-31 -> M=lane-16, same K pattern +8.
__device__ __forceinline__ v16h load_a(const h16* A, int K, int m0, int k0, int lane){
  int m = m0 + (lane & 15);
  const gl_h16* p = (const gl_h16*)(A + (size_t)m*K + k0 + ((lane & 16) ? 8 : 0));
  v16h a;
#pragma unroll
  for (int i=0;i<8;++i){ a[i] = p[i]; a[8+i] = p[16+i]; }
  return a;
}
// B (32x16, KxN) direct from global Wt[K,N]: lane L -> N=n0+L%16; halves i -> K same pattern as A.
__device__ __forceinline__ v16h load_b_g(const h16* Wt, int N, int k0, int n0, int lane){
  const gl_h16* p = (const gl_h16*)(Wt + (size_t)(k0 + ((lane & 16) ? 8 : 0))*N + n0 + (lane & 15));
  v16h b;
#pragma unroll
  for (int i=0;i<8;++i){ b[i] = p[(size_t)i*N]; b[8+i] = p[(size_t)(16+i)*N]; }
  return b;
}
// B from an LDS-staged [KCH x 16] panel (row-compacted) -> ds_load path.
__device__ __forceinline__ v16h load_b_l(const h16* P, int k0, int lane){
  const ls_h16* p = (const ls_h16*)(P + (size_t)(k0 + ((lane & 16) ? 8 : 0))*16 + (lane & 15));
  v16h b;
#pragma unroll
  for (int i=0;i<8;++i){ b[i] = p[i*16]; b[8+i] = p[(16+i)*16]; }
  return b;
}

// TDM issue: stage a [rows x 16] f16 sub-tile of Wt (row stride rowElems) into LDS
// at byte address ldsAddr. D# built per CDNA5 ISA 8.3/8.4; descriptor words forced
// uniform (SGPR) via readfirstlane. 6-arg builtin form on this toolchain.
__device__ __forceinline__ void tdm_issue(const h16* gsrc, int rowElems,
                                          unsigned ldsAddr, int rows){
#if TDM_OK
  unsigned long long ga = (unsigned long long)gsrc;
  v4u g0; v8i g1; v4i gz4 = {0,0,0,0}; v8i gz8 = {0,0,0,0,0,0,0,0};
  g0[0] = rl(1u);                                   // count=1 (valid user D#)
  g0[1] = rl(ldsAddr);                              // lds_addr
  g0[2] = rl((unsigned)ga);                         // global_addr[31:0]
  g0[3] = rl(((unsigned)((ga >> 32) & 0x01FFFFFFull)) | (2u << 30)); // addr[56:32]|type=2
  const unsigned td0 = (unsigned)rowElems;          // tensor_dim0 (elements/row)
  const unsigned td1 = (unsigned)rows;              // tensor_dim1 (rows from base)
  g1[0] = (int)rl(0x00010000u);                     // data_size=1 -> 2 bytes
  g1[1] = (int)rl((td0 & 0xFFFFu) << 16);           // tensor_dim0[15:0]
  g1[2] = (int)rl((td0 >> 16) | ((td1 & 0xFFFFu) << 16)); // dim0 hi | dim1 lo
  g1[3] = (int)rl((td1 >> 16) | (16u << 16));       // dim1 hi | tile_dim0=16
  g1[4] = (int)rl((unsigned)rows);                  // tile_dim1=rows, tile_dim2=0
  g1[5] = (int)rl(td0);                             // tensor_dim0_stride[31:0]
  g1[6] = 0; g1[7] = 0;                             // stride hi, dim1_stride
  __builtin_amdgcn_tensor_load_to_lds(g0, g1, gz4, gz4, gz8, 0);
#else
  (void)gsrc; (void)rowElems; (void)ldsAddr; (void)rows;
#endif
}

// Accumulate one N-tile column panel: acc{0..3} (M-tiles 0,16,32,48) += A[64,K]@Wt[K,16].
// B panels double-buffered through this wave's private 2x2KB LDS region via TDM:
// issue chunk i+1, s_wait_tensorcnt 1, consume chunk i (0 on the last chunk).
__device__ __forceinline__ void panel_mm(const h16* A, int K, const h16* Wt, int NC, int n0,
                                         h16* lbuf, unsigned lbase, int lane,
                                         v8f& c0, v8f& c1, v8f& c2, v8f& c3){
#if TDM_OK
  const int nChunks = K / KCH;
  tdm_issue(Wt + n0, NC, lbase, KCH);
  for (int ci = 0; ci < nChunks; ++ci){
    const int kc = ci*KCH;
    const unsigned cur = (ci & 1) ? PANEL_H : 0u;
    if (ci + 1 < nChunks){
      const unsigned nxt = ((ci + 1) & 1) ? PANEL_H : 0u;
      tdm_issue(Wt + (size_t)(kc + KCH)*NC + n0, NC, lbase + nxt*2u, KCH);
      __builtin_amdgcn_s_wait_tensorcnt((short)1);
    } else {
      __builtin_amdgcn_s_wait_tensorcnt((short)0);
    }
    const h16* cb = lbuf + cur;
#pragma unroll
    for (int k0 = 0; k0 < KCH; k0 += 32){
      v16h b = load_b_l(cb, k0, lane);
      c0 = __builtin_amdgcn_wmma_f32_16x16x32_f16(false, load_a(A,K, 0,kc+k0,lane), false, b, (short)0, c0, false, false);
      c1 = __builtin_amdgcn_wmma_f32_16x16x32_f16(false, load_a(A,K,16,kc+k0,lane), false, b, (short)0, c1, false, false);
      c2 = __builtin_amdgcn_wmma_f32_16x16x32_f16(false, load_a(A,K,32,kc+k0,lane), false, b, (short)0, c2, false, false);
      c3 = __builtin_amdgcn_wmma_f32_16x16x32_f16(false, load_a(A,K,48,kc+k0,lane), false, b, (short)0, c3, false, false);
    }
  }
#else
  for (int kc = 0; kc < K; kc += KCH){
#pragma unroll
    for (int k0 = 0; k0 < KCH; k0 += 32){
      v16h b = load_b_g(Wt, NC, kc+k0, n0, lane);
      c0 = __builtin_amdgcn_wmma_f32_16x16x32_f16(false, load_a(A,K, 0,kc+k0,lane), false, b, (short)0, c0, false, false);
      c1 = __builtin_amdgcn_wmma_f32_16x16x32_f16(false, load_a(A,K,16,kc+k0,lane), false, b, (short)0, c1, false, false);
      c2 = __builtin_amdgcn_wmma_f32_16x16x32_f16(false, load_a(A,K,32,kc+k0,lane), false, b, (short)0, c2, false, false);
      c3 = __builtin_amdgcn_wmma_f32_16x16x32_f16(false, load_a(A,K,48,kc+k0,lane), false, b, (short)0, c3, false, false);
    }
  }
#endif
  (void)lbuf; (void)lbase;
}

// D[64, NC] = A1[64,K1]@Wt1 (+ A2[64,K2]@Wt2) + b1 + b2, opt relu; fp32 out + f16 mirror.
// One N-tile per wave (waveU is readfirstlane'd -> scalar loop, EXEC stays all-1s),
// 4 M-tiles register-blocked so each staged B fragment feeds 4 WMMAs.
__device__ __forceinline__ void gemm_wmma(const h16* A1, int K1, const h16* Wt1,
                          const h16* A2, int K2, const h16* Wt2,
                          const float* b1, const float* b2,
                          float* out, h16* out_h, int NC, bool relu,
                          h16* lbuf, unsigned lbase,
                          int waveU, int lane)
{
  const int nTilesN = NC >> 4;
  for (int nt = waveU; nt < nTilesN; nt += NWAVES){
    const int n0 = nt << 4;
    __builtin_prefetch(A1, 0, 0);   // global_prefetch_b8
    v8f c0 = {}, c1 = {}, c2 = {}, c3 = {};
    panel_mm(A1, K1, Wt1, NC, n0, lbuf, lbase, lane, c0, c1, c2, c3);
    if (A2) panel_mm(A2, K2, Wt2, NC, n0, lbuf, lbase, lane, c0, c1, c2, c3);
    // C/D layout: reg r, lanes 0-15: M=m0+r, N=lane; lanes 16-31: M=m0+8+r, N=lane-16
    const int n = n0 + (lane & 15);
    const float bias = (b1 ? b1[n] : 0.0f) + (b2 ? b2[n] : 0.0f);
    const int mhi = (lane & 16) ? 8 : 0;
    v8f accs[4] = {c0, c1, c2, c3};
#pragma unroll
    for (int mt = 0; mt < 4; ++mt){
#pragma unroll
      for (int r = 0; r < 8; ++r){
        float v = accs[mt][r] + bias;
        if (relu) v = fmaxf(v, 0.0f);
        const size_t idx = (size_t)(mt*16 + mhi + r)*NC + n;
        out[idx] = v;
        if (out_h) out_h[idx] = (h16)v;
      }
    }
  }
}

// ------------------- prep kernels -------------------
__global__ void k_f16t(const float* __restrict__ W, h16* __restrict__ Wt, int N, int K){
  for (long i = (long)blockIdx.x*blockDim.x + threadIdx.x; i < (long)N*K;
       i += (long)gridDim.x*blockDim.x){
    int n = (int)(i / K), k = (int)(i % K);
    Wt[(long)k*N + n] = (h16)W[i];
  }
}

__global__ void k_init(const int* __restrict__ x, const float* __restrict__ emb, int nVocab,
                       float* Bemb, float* s_b, float* V, float* s,
                       float* h, float* c, h16* h_h){
  const long total = (long)Msz*Bsz*Hsz;  // largest array (V)
  for (long i = (long)blockIdx.x*blockDim.x + threadIdx.x; i < total;
       i += (long)gridDim.x*blockDim.x){
    V[i] = 0.0f;
    if (i < (long)Bsz*Ssz*Hsz){
      int bs = (int)(i / Hsz), hh = (int)(i % Hsz);
      int idx = x[bs];
      idx = idx < 0 ? 0 : (idx >= nVocab ? nVocab-1 : idx);
      Bemb[i] = emb[(long)idx*Hsz + hh];
    }
    if (i < (long)Msz*Bsz) s[i] = 0.0f;
    if (i < (long)Bsz*Ssz) s_b[i] = (x[i] > 0) ? 1.0f : 0.0f;
    if (i < (long)Bsz*Hsz){ h[i]=0.0f; c[i]=0.0f; h_h[i]=(h16)0.0f; }
  }
}

// ------------------- persistent scan kernel -------------------
struct SP {
  const float *bih, *bhh, *alpha_w, *alpha_b, *btl, *btr, *l0_b, *l1_b, *l2_w, *l2_b;
  const h16 *WihT, *WhhT, *WtlT, *WtrT, *L0T, *L1T;
  float *Bemb, *s_b, *V, *s, *h, *c;
  h16   *h_h, *xt, *r1h, *r2h, *y0h;
  float *g, *tg, *r1, *r2, *w1g, *w2g, *y1, *out;
};

__global__ __launch_bounds__(512) void spinn_main(SP p){
  __shared__ float wbuf[Bsz*Ssz];            // 12 KB: soft buffer weights
  __shared__ float aR[Bsz], aS[Bsz];
  __shared__ float logit[Bsz*2];
  __shared__ h16  stageB[NWAVES*2*PANEL_H];  // 64 KB: per-wave double-buffered TDM panels
  const int tid    = threadIdx.x;
  const int lane   = tid & 31;
  const int waveU  = (int)rl((unsigned)(tid >> 5));   // scalar wave id
  const int bOf    = tid >> 3;               // (b,h) map: 8 threads per batch row
  const int hOf    = tid & 7;
  h16* lbuf = &stageB[waveU*2*PANEL_H];
  const unsigned lbase = (unsigned)(unsigned long long)(void*)lbuf;

  for (int t = 0; t < TIME; ++t){
    // -- 1a: buffer weights, alpha = 1 --
    if (tid < Bsz){
      float P = 0.0f, prev = 0.0f;
      for (int q = 0; q < Ssz; ++q){
        P += p.s_b[tid*Ssz + q];
        float pc = fminf(P, 1.0f);
        wbuf[tid*Ssz + q] = pc - prev; prev = pc;
      }
    }
    // -- 1b: stack weights, alpha = 1 (peek) --
    if (tid >= 64 && tid < 128){
      const int b = tid - 64;
      float P = 0.0f, p1 = 0.0f, p2 = 0.0f;
      for (int m = 0; m < Msz; ++m){
        P += p.s[m*Bsz + b];
        float c1 = fminf(P, 1.0f), c2 = fminf(P, 2.0f);
        float w1 = c1 - p1;
        p.w1g[m*Bsz + b] = w1;
        p.w2g[m*Bsz + b] = (c2 - p2) - w1;
        p1 = c1; p2 = c2;
      }
    }
    __syncthreads();
    // -- 1c: x_b, x1, x2 -> xt (f16) --
    for (int j = 0; j < 32; ++j){
      const int hh = hOf + 8*j;
      float xb = 0.0f;
      for (int q = 0; q < Ssz; ++q)
        xb += wbuf[bOf*Ssz + q] * p.Bemb[(size_t)(bOf*Ssz + q)*Hsz + hh];
      float a1 = 0.0f, a2 = 0.0f;
      for (int m = 0; m < Msz; ++m){
        float v = p.V[(size_t)(m*Bsz + bOf)*Hsz + hh];
        a1 += p.w1g[m*Bsz + bOf]*v;
        a2 += p.w2g[m*Bsz + bOf]*v;
      }
      p.xt[bOf*XT + hh]         = (h16)xb;
      p.xt[bOf*XT + Hsz + hh]   = (h16)a1;
      p.xt[bOf*XT + 2*Hsz + hh] = (h16)a2;
    }
    __syncthreads();
    // -- 2: LSTM gates: g = xt@Wih^T + h@Whh^T + bih + bhh --
    gemm_wmma(p.xt, XT, p.WihT, p.h_h, Hsz, p.WhhT, p.bih, p.bhh,
              p.g, nullptr, G4, false, lbuf, lbase, waveU, lane);
    __syncthreads();
    // -- 2b: LSTM cell --
    for (int j = 0; j < 32; ++j){
      const int hh = hOf + 8*j;
      float gi = p.g[bOf*G4 + hh];
      float gf = p.g[bOf*G4 + Hsz + hh];
      float gg = p.g[bOf*G4 + 2*Hsz + hh];
      float go = p.g[bOf*G4 + 3*Hsz + hh];
      float cc = sigm(gf)*p.c[bOf*Hsz + hh] + sigm(gi)*tanhf(gg);
      float hn = sigm(go)*tanhf(cc);
      p.c[bOf*Hsz + hh]   = cc;
      p.h[bOf*Hsz + hh]   = hn;
      p.h_h[bOf*Hsz + hh] = (h16)hn;
    }
    __syncthreads();
    // -- 3: alphas = softmax(BETA*(h@alpha_w^T + alpha_b)) --
    if (tid < 128){
      const int b = tid >> 1, jj = tid & 1;
      float d = p.alpha_b[jj];
      for (int k = 0; k < Hsz; ++k) d += p.h[b*Hsz + k]*p.alpha_w[jj*Hsz + k];
      logit[b*2 + jj] = BETA_*d;
    }
    __syncthreads();
    if (tid < Bsz){
      float l0 = logit[tid*2], l1 = logit[tid*2 + 1];
      float mx = fmaxf(l0, l1);
      float e0 = __expf(l0 - mx), e1 = __expf(l1 - mx);
      float inv = 1.0f/(e0 + e1);
      aR[tid] = e0*inv; aS[tid] = e1*inv;
    }
    __syncthreads();
    // -- 4a: stack weights alpha=a_r, pop --
    if (tid < Bsz){
      const int b = tid; const float al = aR[b];
      float P = 0.0f, p1 = 0.0f, p2 = 0.0f;
      for (int m = 0; m < Msz; ++m){
        P += p.s[m*Bsz + b];
        float c1 = fminf(P, al), c2 = fminf(P, 1.0f + al);
        float w1 = c1 - p1, wt = c2 - p2;
        p.w1g[m*Bsz + b] = w1;
        p.w2g[m*Bsz + b] = wt - w1;
        p.s[m*Bsz + b]  -= wt;
        p1 = c1; p2 = c2;
      }
    }
    __syncthreads();
    // -- 4b: r1, r2 --
    for (int j = 0; j < 32; ++j){
      const int hh = hOf + 8*j;
      float a1 = 0.0f, a2 = 0.0f;
      for (int m = 0; m < Msz; ++m){
        float v = p.V[(size_t)(m*Bsz + bOf)*Hsz + hh];
        a1 += p.w1g[m*Bsz + bOf]*v;
        a2 += p.w2g[m*Bsz + bOf]*v;
      }
      p.r1[bOf*Hsz + hh]  = a1;       p.r2[bOf*Hsz + hh]  = a2;
      p.r1h[bOf*Hsz + hh] = (h16)a1;  p.r2h[bOf*Hsz + hh] = (h16)a2;
    }
    __syncthreads();
    // -- 5: tree gates: tg = r1@Wtl^T + r2@Wtr^T + btl + btr --
    gemm_wmma(p.r1h, Hsz, p.WtlT, p.r2h, Hsz, p.WtrT, p.btl, p.btr,
              p.tg, nullptr, G5, false, lbuf, lbase, waveU, lane);
    __syncthreads();
    // -- 6: tree cell; push ht -> V[idx], s[idx] = a_r --
    {
      const int idx = 191 - 2*t;
      for (int j = 0; j < 32; ++j){
        const int hh = hOf + 8*j;
        float ta = p.tg[bOf*G5 + hh];
        float ti = p.tg[bOf*G5 + Hsz + hh];
        float f1 = p.tg[bOf*G5 + 2*Hsz + hh];
        float f2 = p.tg[bOf*G5 + 3*Hsz + hh];
        float to = p.tg[bOf*G5 + 4*Hsz + hh];
        float ct = tanhf(ta)*sigm(ti) + sigm(f1)*p.r1[bOf*Hsz + hh]
                                      + sigm(f2)*p.r2[bOf*Hsz + hh];
        p.V[(size_t)(idx*Bsz + bOf)*Hsz + hh] = sigm(to)*tanhf(ct);
      }
      if (tid < Bsz) p.s[idx*Bsz + tid] = aR[tid];
    }
    __syncthreads();
    // -- 7: buffer pop with a_s; push buf -> V[idx2], s[idx2] = a_s --
    if (tid < Bsz){
      const int b = tid; const float al = aS[b];
      float P = 0.0f, prev = 0.0f;
      for (int q = 0; q < Ssz; ++q){
        P += p.s_b[b*Ssz + q];
        float pc = fminf(P, al);
        float w = pc - prev;
        wbuf[b*Ssz + q] = w;
        p.s_b[b*Ssz + q] -= w;
        prev = pc;
      }
    }
    __syncthreads();
    {
      const int idx2 = 190 - 2*t;
      for (int j = 0; j < 32; ++j){
        const int hh = hOf + 8*j;
        float acc = 0.0f;
        for (int q = 0; q < Ssz; ++q)
          acc += wbuf[bOf*Ssz + q] * p.Bemb[(size_t)(bOf*Ssz + q)*Hsz + hh];
        p.V[(size_t)(idx2*Bsz + bOf)*Hsz + hh] = acc;
      }
      if (tid < Bsz) p.s[idx2*Bsz + tid] = aS[tid];
    }
    __syncthreads();
  }

  // -------- final stack read (ones) + MLP head --------
  if (tid < Bsz){
    const int b = tid;
    float P = 0.0f, p1 = 0.0f;
    for (int m = 0; m < Msz; ++m){
      P += p.s[m*Bsz + b];
      float c1 = fminf(P, 1.0f);
      p.w1g[m*Bsz + b] = c1 - p1; p1 = c1;
    }
  }
  __syncthreads();
  for (int j = 0; j < 32; ++j){
    const int hh = hOf + 8*j;
    float a1 = 0.0f;
    for (int m = 0; m < Msz; ++m)
      a1 += p.w1g[m*Bsz + bOf] * p.V[(size_t)(m*Bsz + bOf)*Hsz + hh];
    p.r1h[bOf*Hsz + hh] = (h16)a1;
  }
  __syncthreads();
  gemm_wmma(p.r1h, Hsz, p.L0T, nullptr, 0, nullptr, p.l0_b, nullptr,
            p.g, p.y0h, MLPD, true, lbuf, lbase, waveU, lane);
  __syncthreads();
  gemm_wmma(p.y0h, MLPD, p.L1T, nullptr, 0, nullptr, p.l1_b, nullptr,
            p.y1, nullptr, MLPD, true, lbuf, lbase, waveU, lane);
  __syncthreads();
  if (tid < Bsz*3){
    const int b = tid/3, cls = tid%3;
    float d = p.l2_b[cls];
    for (int k = 0; k < MLPD; ++k) d += p.y1[b*MLPD + k]*p.l2_w[cls*MLPD + k];
    p.out[b*3 + cls] = d;
  }
}

extern "C" void kernel_launch(void* const* d_in, const int* in_sizes, int n_in,
                              void* d_out, int out_size, void* d_ws, size_t ws_size,
                              hipStream_t stream) {
  const int*   x        = (const int*)  d_in[0];
  const float* emb      = (const float*)d_in[1];
  const float* wih      = (const float*)d_in[2];
  const float* whh      = (const float*)d_in[3];
  const float* bih      = (const float*)d_in[4];
  const float* bhh      = (const float*)d_in[5];
  const float* alpha_w  = (const float*)d_in[6];
  const float* alpha_b  = (const float*)d_in[7];
  const float* tlw      = (const float*)d_in[8];
  const float* tlb      = (const float*)d_in[9];
  const float* trw      = (const float*)d_in[10];
  const float* trb      = (const float*)d_in[11];
  const float* l0w      = (const float*)d_in[12];
  const float* l0b      = (const float*)d_in[13];
  const float* l1w      = (const float*)d_in[14];
  const float* l1b      = (const float*)d_in[15];
  const float* l2w      = (const float*)d_in[16];
  const float* l2b      = (const float*)d_in[17];
  const int nVocab = in_sizes[1] / Hsz;

  char* ws = (char*)d_ws;
  size_t off = 0;
  auto alloc = [&](size_t bytes)->char* {
    off = (off + 255) & ~(size_t)255;
    char* ptr = ws + off;
    off += bytes;
    return ptr;
  };
  h16*  WihT = (h16*)alloc((size_t)XT  * G4  * 2);
  h16*  WhhT = (h16*)alloc((size_t)Hsz * G4  * 2);
  h16*  WtlT = (h16*)alloc((size_t)Hsz * G5  * 2);
  h16*  WtrT = (h16*)alloc((size_t)Hsz * G5  * 2);
  h16*  L0T  = (h16*)alloc((size_t)Hsz * MLPD* 2);
  h16*  L1T  = (h16*)alloc((size_t)MLPD* MLPD* 2);
  float* Bemb= (float*)alloc((size_t)Bsz*Ssz*Hsz*4);
  float* s_b = (float*)alloc((size_t)Bsz*Ssz*4);
  float* V   = (float*)alloc((size_t)Msz*Bsz*Hsz*4);
  float* s   = (float*)alloc((size_t)Msz*Bsz*4);
  float* h   = (float*)alloc((size_t)Bsz*Hsz*4);
  float* c   = (float*)alloc((size_t)Bsz*Hsz*4);
  h16*  h_h  = (h16*)alloc((size_t)Bsz*Hsz*2);
  h16*  xt   = (h16*)alloc((size_t)Bsz*XT*2);
  float* g   = (float*)alloc((size_t)Bsz*G4*4);
  float* tg  = (float*)alloc((size_t)Bsz*G5*4);
  float* r1  = (float*)alloc((size_t)Bsz*Hsz*4);
  float* r2  = (float*)alloc((size_t)Bsz*Hsz*4);
  h16*  r1h  = (h16*)alloc((size_t)Bsz*Hsz*2);
  h16*  r2h  = (h16*)alloc((size_t)Bsz*Hsz*2);
  float* w1g = (float*)alloc((size_t)Msz*Bsz*4);
  float* w2g = (float*)alloc((size_t)Msz*Bsz*4);
  float* y1  = (float*)alloc((size_t)Bsz*MLPD*4);
  h16*  y0h  = (h16*)alloc((size_t)Bsz*MLPD*2);

  // Weight convert + transpose (fp32 [N,K] -> f16 [K,N]) — L2-resident thereafter.
  k_f16t<<<1024, 256, 0, stream>>>(wih, WihT, G4,  XT);
  k_f16t<<<1024, 256, 0, stream>>>(whh, WhhT, G4,  Hsz);
  k_f16t<<<1024, 256, 0, stream>>>(tlw, WtlT, G5,  Hsz);
  k_f16t<<<1024, 256, 0, stream>>>(trw, WtrT, G5,  Hsz);
  k_f16t<<<1024, 256, 0, stream>>>(l0w, L0T,  MLPD,Hsz);
  k_f16t<<<2048, 256, 0, stream>>>(l1w, L1T,  MLPD,MLPD);
  // State init: embedding gather, s_b = (x>0), zero V/s/h/c.
  k_init<<<2048, 256, 0, stream>>>(x, emb, nVocab, Bemb, s_b, V, s, h, c, h_h);

  SP p;
  p.bih=bih; p.bhh=bhh; p.alpha_w=alpha_w; p.alpha_b=alpha_b;
  p.btl=tlb; p.btr=trb; p.l0_b=l0b; p.l1_b=l1b; p.l2_w=l2w; p.l2_b=l2b;
  p.WihT=WihT; p.WhhT=WhhT; p.WtlT=WtlT; p.WtrT=WtrT; p.L0T=L0T; p.L1T=L1T;
  p.Bemb=Bemb; p.s_b=s_b; p.V=V; p.s=s; p.h=h; p.c=c;
  p.h_h=h_h; p.xt=xt; p.r1h=r1h; p.r2h=r2h; p.y0h=y0h;
  p.g=g; p.tg=tg; p.r1=r1; p.r2=r2; p.w1g=w1g; p.w2g=w2g; p.y1=y1;
  p.out=(float*)d_out;

  spinn_main<<<1, 512, 0, stream>>>(p);
}
